// cosine_layer_54425825575210
// MI455X (gfx1250) — matrix operand order repaired and checked by reading, not compiled, not run on hardware
//
#include <hip/hip_runtime.h>
#include <hip/hip_bf16.h>

#define IN_DIM 4096
#define OUT_DIM 32768
#define BATCH 4096
#define EPSV 1e-8f

#define BM 128
#define BN 128
#define BK 64
#define SA 72   // A tile LDS row stride (bf16 elems): 16B-aligned rows, conflict-free b128 frag loads
#define SW 70   // W^T tile LDS row stride (bf16 elems): conflict-free dword frag loads

typedef __bf16 bf16_t;
typedef bf16_t v16bf __attribute__((ext_vector_type(16)));
typedef bf16_t v2bf  __attribute__((ext_vector_type(2)));
typedef float v8f __attribute__((ext_vector_type(8)));
typedef float v2f __attribute__((ext_vector_type(2)));
typedef unsigned int uint32;
typedef uint32 v4u __attribute__((ext_vector_type(4)));

// native packed f32 -> bf16 conversion (RNE) -> v_cvt_pk_bf16_f32
__device__ __forceinline__ uint32 pk2bf(float lo, float hi) {
  union { v2bf b; uint32 u; } t;
  t.b = __builtin_convertvector((v2f){lo, hi}, v2bf);
  return t.u;
}

// ---------- norm kernels (full f32 precision) ----------

__global__ void cos_xnorm_kernel(const float* __restrict__ x, float* __restrict__ rx) {
  __shared__ float red[256];
  const int row = blockIdx.x;
  const float* p = x + (size_t)row * IN_DIM;
  float s = 0.f;
  #pragma unroll 4
  for (int i = threadIdx.x; i < IN_DIM; i += 256) { float v = p[i]; s += v * v; }
  red[threadIdx.x] = s;
  __syncthreads();
  for (int off = 128; off > 0; off >>= 1) {
    if ((int)threadIdx.x < off) red[threadIdx.x] += red[threadIdx.x + off];
    __syncthreads();
  }
  if (threadIdx.x == 0) rx[row] = 1.f / fmaxf(sqrtf(red[0]), EPSV);
}

__global__ void cos_wnorm_kernel(const float* __restrict__ w, float* __restrict__ rw) {
  const int n = blockIdx.x * 256 + threadIdx.x;
  const float* p = w + n;
  float s = 0.f;
  #pragma unroll 8
  for (int k = 0; k < IN_DIM; ++k) { float v = p[(size_t)k * OUT_DIM]; s += v * v; }
  rw[n] = 1.f / fmaxf(sqrtf(s), EPSV);
}

// ---------- bf16 WMMA GEMM + cosine epilogue, double-buffered LDS, BK=64 ----------

__launch_bounds__(256, 2)
__global__ void cos_gemm_kernel(const float* __restrict__ x,
                                const float* __restrict__ w,
                                const float* __restrict__ rx,
                                const float* __restrict__ rw,
                                float* __restrict__ out) {
  __shared__ __align__(16) unsigned short sA[2][BM * SA];
  __shared__ __align__(16) unsigned short sW[2][BN * SW];

  const int tid  = threadIdx.x;
  const int lane = tid & 31;
  const int wave = tid >> 5;
  const int m0 = blockIdx.x * BM;   // M fastest-varying: consecutive blocks share W N-panel (L2 reuse)
  const int n0 = blockIdx.y * BN;
  const int waveM = (wave & 1) * 64;   // 2 waves along M
  const int waveN = (wave >> 1) * 32;  // 4 waves along N

  // A-fill mapping: thread -> (row, half of K): 32 floats each
  const int arow = tid >> 1;            // 0..127
  const int akh  = (tid & 1) * 32;      // 0 or 32 (elems)
  // W-fill mapping: thread -> (k pair in each 32-half, 8 consecutive n)
  const int wkp  = tid >> 4;            // 0..15 -> k = 2*wkp, 2*wkp+1 (+32 for 2nd half)
  const int wnp  = tid & 15;            // -> n = 8*wnp .. 8*wnp+7

  // per-lane fragment offsets (A and B both use contiguous k = 16*hi + s order)
  const int hi   = lane >> 4;           // 0/1: lane half selects K sub-range
  const int aK0  = hi * 16;             // A chunk0 elem offset; chunk1 at +8
  const int bK0  = hi * 16;             // B 32B chunk elem offset
  const int colL = lane & 15;
  const int rowSel = hi * 8;            // C: rows hi*8+v

  v8f acc[4][2];
  #pragma unroll
  for (int i = 0; i < 4; ++i)
    #pragma unroll
    for (int j = 0; j < 2; ++j)
      acc[i][j] = (v8f){0.f, 0.f, 0.f, 0.f, 0.f, 0.f, 0.f, 0.f};

  const float* gA = x + (size_t)(m0 + arow) * IN_DIM + akh;
  const float* gW = w + (size_t)(2 * wkp) * OUT_DIM + n0 + 8 * wnp;

  // fragment / fill LDS element offsets (kept as ints so addresses stay in LDS addrspace)
  int aOff[4], bOff[2];
  #pragma unroll
  for (int i = 0; i < 4; ++i) aOff[i] = (waveM + 16 * i + colL) * SA + aK0;
  #pragma unroll
  for (int j = 0; j < 2; ++j) bOff[j] = (waveN + 16 * j + colL) * SW + bK0;
  const int aStoreOff = arow * SA + akh;
  const int wStoreOff = (8 * wnp) * SW + 2 * wkp;

  // ---- helpers (always called with literal buf -> loop-invariant LDS addresses) ----
  auto loadG = [&](int k0, float4* aR, float4* wR) {
    const float4* pa = (const float4*)(gA + k0);
    #pragma unroll
    for (int q = 0; q < 8; ++q) aR[q] = pa[q];            // 32 contiguous floats of one x row
    const float* pw = gW + (size_t)k0 * OUT_DIM;
    #pragma unroll
    for (int h = 0; h < 2; ++h) {                         // k halves: rows 2wkp(+1), 32+2wkp(+1)
      const float* ph = pw + (size_t)(32 * h) * OUT_DIM;
      wR[4 * h + 0] = *(const float4*)(ph);
      wR[4 * h + 1] = *(const float4*)(ph + 4);
      wR[4 * h + 2] = *(const float4*)(ph + OUT_DIM);
      wR[4 * h + 3] = *(const float4*)(ph + OUT_DIM + 4);
    }
  };
  auto storeL = [&](int buf, const float4* aR, const float4* wR) {
    uint32* da = (uint32*)&sA[buf][aStoreOff];
    #pragma unroll
    for (int q = 0; q < 8; ++q) {
      da[2 * q]     = pk2bf(aR[q].x, aR[q].y);
      da[2 * q + 1] = pk2bf(aR[q].z, aR[q].w);
    }
    #pragma unroll
    for (int h = 0; h < 2; ++h) {
      uint32* dw = (uint32*)&sW[buf][wStoreOff + 32 * h];
      const float4 a = wR[4*h+0], b = wR[4*h+1], c = wR[4*h+2], d = wR[4*h+3];
      dw[0 * (SW/2)] = pk2bf(a.x, c.x);
      dw[1 * (SW/2)] = pk2bf(a.y, c.y);
      dw[2 * (SW/2)] = pk2bf(a.z, c.z);
      dw[3 * (SW/2)] = pk2bf(a.w, c.w);
      dw[4 * (SW/2)] = pk2bf(b.x, d.x);
      dw[5 * (SW/2)] = pk2bf(b.y, d.y);
      dw[6 * (SW/2)] = pk2bf(b.z, d.z);
      dw[7 * (SW/2)] = pk2bf(b.w, d.w);
    }
  };
  auto compute = [&](int buf) {
    #pragma unroll
    for (int s = 0; s < 2; ++s) {                         // two 32-K WMMA steps per stage
      union AF { v16bf v; v4u u[2]; } af[4];
      union BF { v16bf v; uint32 u[8]; } bfr[2];
      #pragma unroll
      for (int i = 0; i < 4; ++i) {
        const unsigned short* base = &sA[buf][aOff[i] + 32 * s];
        af[i].u[0] = *(const v4u*)(base);
        af[i].u[1] = *(const v4u*)(base + 8);
      }
      #pragma unroll
      for (int j = 0; j < 2; ++j) {
        const uint32* p = (const uint32*)&sW[buf][bOff[j] + 32 * s];
        #pragma unroll
        for (int q = 0; q < 8; ++q) bfr[j].u[q] = p[q];
      }
      #pragma unroll
      for (int i = 0; i < 4; ++i)
        #pragma unroll
        for (int j = 0; j < 2; ++j)
          acc[i][j] = __builtin_amdgcn_wmma_f32_16x16x32_bf16(
              false, af[i].v, false, bfr[j].v, (short)0, acc[i][j], false, false);
    }
  };

  // ---- software pipeline, buffers as compile-time constants ----
  {
    float4 aR[8], wR[8];
    loadG(0, aR, wR);
    storeL(0, aR, wR);
  }
  __syncthreads();

  int k0 = BK;
  #pragma unroll 1
  for (int r = 0; r < (IN_DIM / BK - 2) / 2; ++r) {       // 31 double-rounds
    {
      float4 aR[8], wR[8];
      loadG(k0, aR, wR);        // global loads issue early, overlap 16 WMMAs
      compute(0);
      storeL(1, aR, wR);
      __syncthreads();
      k0 += BK;
    }
    {
      float4 aR[8], wR[8];
      loadG(k0, aR, wR);
      compute(1);
      storeL(0, aR, wR);
      __syncthreads();
      k0 += BK;
    }
  }
  {
    float4 aR[8], wR[8];
    loadG(k0, aR, wR);          // stage 63
    compute(0);                 // stage 62
    storeL(1, aR, wR);
    __syncthreads();
  }
  compute(1);                   // stage 63

  // ---- cosine epilogue: scale by reciprocal norms, streaming stores ----
  const int nBase = n0 + waveN + colL;
  const float rw0 = rw[nBase];
  const float rw1 = rw[nBase + 16];
  #pragma unroll
  for (int i = 0; i < 4; ++i) {
    const int rbase = m0 + waveM + 16 * i + rowSel;
    #pragma unroll
    for (int v = 0; v < 8; ++v) {
      const float rxv = rx[rbase + v];
      float* po = out + (size_t)(rbase + v) * OUT_DIM + nBase;
      __builtin_nontemporal_store(acc[i][0][v] * rxv * rw0, po);
      __builtin_nontemporal_store(acc[i][1][v] * rxv * rw1, po + 16);
    }
  }
}

extern "C" void kernel_launch(void* const* d_in, const int* in_sizes, int n_in,
                              void* d_out, int out_size, void* d_ws, size_t ws_size,
                              hipStream_t stream) {
  const float* x = (const float*)d_in[0];
  const float* w = (const float*)d_in[1];
  float* out = (float*)d_out;
  float* rx = (float*)d_ws;          // BATCH reciprocal row norms
  float* rw = rx + BATCH;            // OUT_DIM reciprocal col norms

  cos_xnorm_kernel<<<BATCH, 256, 0, stream>>>(x, rx);
  cos_wnorm_kernel<<<OUT_DIM / 256, 256, 0, stream>>>(w, rw);
  dim3 grid(BATCH / BM, OUT_DIM / BN);   // M-tiles fastest -> W panel reuse in L2
  cos_gemm_kernel<<<grid, 256, 0, stream>>>(x, w, rx, rw, out);
}
